// ActionScore_6227702579943
// MI455X (gfx1250) — matrix operand order repaired and checked by reading, not compiled, not run on hardware
//
#include <hip/hip_runtime.h>
#include <hip/hip_bf16.h>
#include <math.h>

typedef __attribute__((ext_vector_type(16))) _Float16 v16h;
typedef __attribute__((ext_vector_type(8)))  _Float16 v8h;
typedef __attribute__((ext_vector_type(8)))  float    v8f;
typedef __attribute__((address_space(3)))    _Float16 lds_f16;

#define EPS_BN 1e-5f
#define NB 16          // batch
#define NPTS 4096      // input points

// ---------------------------------------------------------------------------
// Weight prep: f32 (o x c) -> f16 (o x kpad, zero padded), BN folded scale/shift
// ---------------------------------------------------------------------------
__global__ void prep_weights_kernel(const float* __restrict__ W, const float* __restrict__ ga,
                                    const float* __restrict__ be, const float* __restrict__ mu,
                                    const float* __restrict__ va,
                                    _Float16* __restrict__ Wh, float* __restrict__ scale,
                                    float* __restrict__ shift, int o, int c, int kpad) {
  int tid = blockIdx.x * blockDim.x + threadIdx.x;
  int total = o * kpad;
  if (tid < total) {
    int oo = tid / kpad, cc = tid % kpad;
    Wh[tid] = (cc < c) ? (_Float16)W[oo * c + cc] : (_Float16)0.f;
  }
  if (tid < o) {
    float s = ga[tid] * rsqrtf(va[tid] + EPS_BN);
    scale[tid] = s;
    shift[tid] = be[tid] - mu[tid] * s;
  }
}

__global__ void f32_to_f16_kernel(const float* __restrict__ in, _Float16* __restrict__ out, int n) {
  int tid = blockIdx.x * blockDim.x + threadIdx.x;
  if (tid < n) out[tid] = (_Float16)in[tid];
}

// ---------------------------------------------------------------------------
// Farthest point sampling: one block per batch, dist[] in LDS, serial npoint loop
// ---------------------------------------------------------------------------
__global__ void fps_kernel(const float* __restrict__ xyz, int n, int npoint, int* __restrict__ out) {
  const int b = blockIdx.x, t = threadIdx.x;
  __shared__ float dist[4096];
  __shared__ float rv[256];
  __shared__ int   ri[256];
  __shared__ int   sfar;
  const float* X = xyz + (size_t)b * n * 3;
  for (int p = t; p < n; p += 256) dist[p] = 1e10f;
  if (t == 0) sfar = 0;
  __syncthreads();
  for (int it = 0; it < npoint; ++it) {
    int far = sfar;
    if (t == 0) out[b * npoint + it] = far;
    float cx = X[far * 3], cy = X[far * 3 + 1], cz = X[far * 3 + 2];
    float bv = -1.f; int bi = 0x7fffffff;
    for (int p = t; p < n; p += 256) {
      float dx = X[p * 3] - cx, dy = X[p * 3 + 1] - cy, dz = X[p * 3 + 2] - cz;
      float d = dx * dx + dy * dy + dz * dz;
      float od = dist[p];
      float nd = d < od ? d : od;
      dist[p] = nd;
      if (nd > bv) { bv = nd; bi = p; }
    }
    rv[t] = bv; ri[t] = bi;
    __syncthreads();
    for (int s = 128; s > 0; s >>= 1) {
      if (t < s) {
        if (rv[t + s] > rv[t] || (rv[t + s] == rv[t] && ri[t + s] < ri[t])) {
          rv[t] = rv[t + s]; ri[t] = ri[t + s];
        }
      }
      __syncthreads();
    }
    if (t == 0) sfar = ri[0];
    __syncthreads();
  }
}

__global__ void gather_xyz_kernel(const float* __restrict__ xyz, const int* __restrict__ idx,
                                  int n, int m, int total, float* __restrict__ out) {
  int tid = blockIdx.x * blockDim.x + threadIdx.x;
  if (tid >= total) return;               // total = NB*m
  int b = tid / m;
  int c = idx[tid];
  const float* p = xyz + ((size_t)b * n + c) * 3;
  out[(size_t)tid * 3 + 0] = p[0];
  out[(size_t)tid * 3 + 1] = p[1];
  out[(size_t)tid * 3 + 2] = p[2];
}

// ---------------------------------------------------------------------------
// Ball query: first nsample in-radius indices in ascending order, pad with first
// ---------------------------------------------------------------------------
__global__ void ball_query_kernel(const float* __restrict__ xyz, const float* __restrict__ nxyz,
                                  int n, int m, int ns, float r2, int total, int* __restrict__ idx) {
  int tid = blockIdx.x * blockDim.x + threadIdx.x;
  if (tid >= total) return;               // total = NB*m
  int b = tid / m;
  const float* q = nxyz + (size_t)tid * 3;
  const float* X = xyz + (size_t)b * n * 3;
  int* o = idx + (size_t)tid * ns;
  float qx = q[0], qy = q[1], qz = q[2];
  int cnt = 0, first = 0;
  for (int j = 0; j < n; ++j) {
    float dx = X[j * 3] - qx, dy = X[j * 3 + 1] - qy, dz = X[j * 3 + 2] - qz;
    float d = dx * dx + dy * dy + dz * dz;
    if (d < r2) {
      if (cnt == 0) first = j;
      if (cnt < ns) o[cnt] = j;
      ++cnt;
    }
  }
  int have = cnt < ns ? cnt : ns;
  for (int k = have; k < ns; ++k) o[k] = first;   // first==0 when cnt==0 -> matches ref
}

// ---------------------------------------------------------------------------
// Grouping: A[row, :] = [xyz_c - center (3), feat_c (C), zeros..Kpad] as f16
// ---------------------------------------------------------------------------
__global__ void group_kernel(const float* __restrict__ xyz, const float* __restrict__ nxyz,
                             const _Float16* __restrict__ feat, int C,
                             const int* __restrict__ idx, int m, int ns, int n, int Kpad,
                             int totalRows, _Float16* __restrict__ A) {
  int r = blockIdx.x * blockDim.x + threadIdx.x;
  if (r >= totalRows) return;             // totalRows = NB*m*ns
  int b = r / (m * ns);
  int rem = r % (m * ns);
  int s = rem / ns;
  int c = idx[r];
  const float* p = xyz + ((size_t)b * n + c) * 3;
  const float* q = nxyz + ((size_t)b * m + s) * 3;
  _Float16* out = A + (size_t)r * Kpad;
  out[0] = (_Float16)(p[0] - q[0]);
  out[1] = (_Float16)(p[1] - q[1]);
  out[2] = (_Float16)(p[2] - q[2]);
  const _Float16* f = feat + ((size_t)b * n + c) * C;
  for (int j = 0; j < C; ++j) out[3 + j] = f[j];
  for (int j = 3 + C; j < Kpad; ++j) out[j] = (_Float16)0.f;
}

// ---------------------------------------------------------------------------
// WMMA GEMM + BN + ReLU:  Out[M,O] = relu(A[M,Kpad] x W[O,Kpad]^T * scale + shift)
// block = 128 threads (4 waves); blockIdx.x = 32-row strip; each wave computes
// two 16x16 D tiles (rows 0-15 and 16-31) per O-tile, so each B fragment feeds
// two v_wmma ops. A strip staged to LDS with global_load_async_to_lds_b128
// (ASYNCcnt path), then s_wait_asynccnt + barrier.
// ---------------------------------------------------------------------------
__global__ void __launch_bounds__(128)
gemm_bn_relu_kernel(const _Float16* __restrict__ A, int Kpad, int O,
                    const _Float16* __restrict__ W, const float* __restrict__ scale,
                    const float* __restrict__ shift, _Float16* __restrict__ Out) {
  __shared__ __align__(64) _Float16 at[32 * 768];      // 48KB, max Kpad = 768
  const int tid = threadIdx.x;
  const size_t rowBase = (size_t)blockIdx.x * 32;
  // ---- async stage of the 32 x Kpad A strip into LDS (16B per lane per op) ----
  const int vecs = (32 * Kpad) >> 3;                   // # of 16B chunks
  {
    unsigned ldsBase = (unsigned)(size_t)(lds_f16*)at; // LDS byte offset of `at`
    unsigned long long gbase = (unsigned long long)(size_t)(A + rowBase * Kpad);
    for (int i = tid; i < vecs; i += 128) {
      unsigned laddr = ldsBase + (unsigned)i * 16u;
      unsigned long long gaddr = gbase + (unsigned long long)i * 16ull;
      asm volatile("global_load_async_to_lds_b128 %0, %1, off"
                   :: "v"(laddr), "v"(gaddr) : "memory");
    }
    asm volatile("s_wait_asynccnt 0" ::: "memory");
  }
  __syncthreads();

  const int wave = tid >> 5, lane = tid & 31;
  const int tileO = blockIdx.y * 4 + wave;
  if (tileO * 16 >= O) return;
  const int half = lane >> 4;       // 0: lanes 0-15, 1: lanes 16-31
  const int m = lane & 15;

  v8f acc0 = {};                    // rows rowBase + 0..15
  v8f acc1 = {};                    // rows rowBase + 16..31
  const _Float16* wbase = W + (size_t)(tileO * 16 + m) * Kpad + half * 8;
  const _Float16* arow0 = at + m * Kpad + half * 8;
  const _Float16* arow1 = at + (16 + m) * Kpad + half * 8;
  for (int k0 = 0; k0 < Kpad; k0 += 32) {
    // A fragment per 16-bit 16x32 layout: lane<16 -> K {k0..k0+7, k0+16..k0+23}
    //                                     lane>=16 -> K {k0+8..k0+15, k0+24..k0+31}
    v8h lo0 = *(const v8h*)(arow0 + k0);
    v8h hi0 = *(const v8h*)(arow0 + k0 + 16);
    v8h lo1 = *(const v8h*)(arow1 + k0);
    v8h hi1 = *(const v8h*)(arow1 + k0 + 16);
    v16h a0, a1;
#pragma unroll
    for (int i = 0; i < 8; ++i) {
      a0[i] = lo0[i]; a0[i + 8] = hi0[i];
      a1[i] = lo1[i]; a1[i + 8] = hi1[i];
    }
    // B fragment: lane column n = lane&15 (= W row o), K same order as A fragment
    v8h blo = *(const v8h*)(wbase + k0);
    v8h bhi = *(const v8h*)(wbase + k0 + 16);
    v16h bfrag;
#pragma unroll
    for (int i = 0; i < 8; ++i) { bfrag[i] = blo[i]; bfrag[i + 8] = bhi[i]; }
    acc0 = __builtin_amdgcn_wmma_f32_16x16x32_f16(false, a0, false, bfrag,
                                                  (short)0, acc0, false, false);
    acc1 = __builtin_amdgcn_wmma_f32_16x16x32_f16(false, a1, false, bfrag,
                                                  (short)0, acc1, false, false);
  }
  const int o = tileO * 16 + m;      // D: col n = lane&15
  const float sc = scale[o], sh = shift[o];
#pragma unroll
  for (int j = 0; j < 8; ++j) {
    int mr = j + half * 8;           // D: row = VGPR j (+8 for lanes 16-31)
    float v0 = acc0[j] * sc + sh;
    v0 = v0 > 0.f ? v0 : 0.f;
    Out[(rowBase + mr) * (size_t)O + o] = (_Float16)v0;
    float v1 = acc1[j] * sc + sh;
    v1 = v1 > 0.f ? v1 : 0.f;
    Out[(rowBase + 16 + mr) * (size_t)O + o] = (_Float16)v1;
  }
}

// ---------------------------------------------------------------------------
// Max-pool over nsample
// ---------------------------------------------------------------------------
__global__ void maxpool_kernel(const _Float16* __restrict__ in, int ns, int O, int total,
                               _Float16* __restrict__ out) {
  int tid = blockIdx.x * blockDim.x + threadIdx.x;
  if (tid >= total) return;               // total = points*O
  int p = tid / O, o = tid % O;
  const _Float16* base = in + ((size_t)p * ns) * O + o;
  float mx = -1e30f;
  for (int k = 0; k < ns; ++k) {
    float v = (float)base[(size_t)k * O];
    mx = v > mx ? v : mx;
  }
  out[tid] = (_Float16)mx;
}

// ---------------------------------------------------------------------------
// 3-NN with inverse-distance weights
// ---------------------------------------------------------------------------
__global__ void knn3_kernel(const float* __restrict__ xyz1, const float* __restrict__ xyz2,
                            int n1, int n2, int total, int* __restrict__ idx3,
                            float* __restrict__ w3) {
  int tid = blockIdx.x * blockDim.x + threadIdx.x;
  if (tid >= total) return;               // total = NB*n1
  int b = tid / n1;
  const float* p = xyz1 + (size_t)tid * 3;
  const float* X = xyz2 + (size_t)b * n2 * 3;
  float px = p[0], py = p[1], pz = p[2];
  float d0 = 1e30f, d1 = 1e30f, d2 = 1e30f;
  int i0 = 0, i1 = 0, i2 = 0;
  for (int j = 0; j < n2; ++j) {
    float dx = X[j * 3] - px, dy = X[j * 3 + 1] - py, dz = X[j * 3 + 2] - pz;
    float d = dx * dx + dy * dy + dz * dz;
    if (d < d0)      { d2 = d1; i2 = i1; d1 = d0; i1 = i0; d0 = d; i0 = j; }
    else if (d < d1) { d2 = d1; i2 = i1; d1 = d; i1 = j; }
    else if (d < d2) { d2 = d; i2 = j; }
  }
  float w0 = 1.f / (d0 + 1e-8f), w1 = 1.f / (d1 + 1e-8f), w2 = 1.f / (d2 + 1e-8f);
  float s = w0 + w1 + w2;
  idx3[tid * 3 + 0] = i0; idx3[tid * 3 + 1] = i1; idx3[tid * 3 + 2] = i2;
  w3[tid * 3 + 0] = w0 / s; w3[tid * 3 + 1] = w1 / s; w3[tid * 3 + 2] = w2 / s;
}

// H[row, :] = [sum_t w*f2[idx_t] (C2), f1[row] (C1), zeros..Kpad]  (f16)
__global__ void fillH_kernel(const int* __restrict__ idx3, const float* __restrict__ w3,
                             const _Float16* __restrict__ f2, int C2, int n2,
                             const _Float16* __restrict__ f1, int C1, int n1, int Kpad,
                             int total, _Float16* __restrict__ H) {
  int tid = blockIdx.x * blockDim.x + threadIdx.x;
  if (tid >= total) return;               // total = NB*n1*Kpad
  int row = tid / Kpad, c = tid % Kpad;
  int b = row / n1;
  float v;
  if (c < C2) {
    float acc = 0.f;
#pragma unroll
    for (int t = 0; t < 3; ++t) {
      int j = idx3[row * 3 + t];
      acc += w3[row * 3 + t] * (float)f2[((size_t)b * n2 + j) * C2 + c];
    }
    v = acc;
  } else if (c < C2 + C1) {
    v = (float)f1[(size_t)row * C1 + (c - C2)];
  } else {
    v = 0.f;
  }
  H[(size_t)row * Kpad + c] = (_Float16)v;
}

// ---------------------------------------------------------------------------
// Head: only point 0 per batch is needed by the output.
// ---------------------------------------------------------------------------
struct HeadArgs {
  const float* pcs; const float* task; const float* cp;
  const float* lx1;            // (NB,1024,3)
  const _Float16* f2;          // fp_feat1 (NB,1024,128)
  const float* W0; const float* ga0; const float* be0; const float* mu0; const float* va0; // 128x131
  const float* W1; const float* ga1; const float* be1; const float* mu1; const float* va1; // 128x128
  const float* W2; const float* ga2; const float* be2; const float* mu2; const float* va2; // 128x128
  const float* Wf; const float* gaf; const float* bef; const float* muf; const float* vaf; // fc 128x128
  const float* task_w; const float* task_b; const float* cp_w; const float* cp_b;
  const float* w1; const float* b1; const float* w2; const float* b2; const float* w3; const float* b3;
  float* out;
};

__device__ inline void dense_bn_relu(const float* in, float* out, const float* W,
                                     const float* ga, const float* be, const float* mu,
                                     const float* va, int cin, int t) {
  if (t < 128) {
    const float* w = W + (size_t)t * cin;
    float acc = 0.f;
    for (int c = 0; c < cin; ++c) acc += w[c] * in[c];
    float s = ga[t] * rsqrtf(va[t] + EPS_BN);
    float v = (acc - mu[t]) * s + be[t];
    out[t] = v > 0.f ? v : 0.f;
  }
}

__global__ void __launch_bounds__(128) head_kernel(HeadArgs A) {
  const int b = blockIdx.x, t = threadIdx.x;
  __shared__ float bufA[192], bufB[192];
  __shared__ float kw[3];
  __shared__ int   ki[3];
  if (t == 0) {
    const float* p = A.pcs + (size_t)b * NPTS * 3;   // point 0
    float px = p[0], py = p[1], pz = p[2];
    const float* X = A.lx1 + (size_t)b * 1024 * 3;
    float d0 = 1e30f, d1 = 1e30f, d2 = 1e30f;
    int i0 = 0, i1 = 0, i2 = 0;
    for (int j = 0; j < 1024; ++j) {
      float dx = X[j * 3] - px, dy = X[j * 3 + 1] - py, dz = X[j * 3 + 2] - pz;
      float d = dx * dx + dy * dy + dz * dz;
      if (d < d0)      { d2 = d1; i2 = i1; d1 = d0; i1 = i0; d0 = d; i0 = j; }
      else if (d < d1) { d2 = d1; i2 = i1; d1 = d; i1 = j; }
      else if (d < d2) { d2 = d; i2 = j; }
    }
    float w0 = 1.f / (d0 + 1e-8f), w1 = 1.f / (d1 + 1e-8f), w2 = 1.f / (d2 + 1e-8f);
    float s = w0 + w1 + w2;
    kw[0] = w0 / s; kw[1] = w1 / s; kw[2] = w2 / s;
    ki[0] = i0; ki[1] = i1; ki[2] = i2;
  }
  __syncthreads();
  // h = [interp (128), f1 = pcs[b,0,:] (3)]
  if (t < 128) {
    const _Float16* F = A.f2 + (size_t)b * 1024 * 128;
    float v = kw[0] * (float)F[(size_t)ki[0] * 128 + t]
            + kw[1] * (float)F[(size_t)ki[1] * 128 + t]
            + kw[2] * (float)F[(size_t)ki[2] * 128 + t];
    bufA[t] = v;
  }
  if (t < 3) bufA[128 + t] = A.pcs[(size_t)b * NPTS * 3 + t];
  __syncthreads();
  dense_bn_relu(bufA, bufB, A.W0, A.ga0, A.be0, A.mu0, A.va0, 131, t); __syncthreads();
  dense_bn_relu(bufB, bufA, A.W1, A.ga1, A.be1, A.mu1, A.va1, 128, t); __syncthreads();
  dense_bn_relu(bufA, bufB, A.W2, A.ga2, A.be2, A.mu2, A.va2, 128, t); __syncthreads();
  dense_bn_relu(bufB, bufA, A.Wf, A.gaf, A.bef, A.muf, A.vaf, 128, t); // net -> bufA[0:128]
  if (t < 32) {
    bufA[128 + t] = A.task[b] * A.task_w[t] + A.task_b[t];
    const float* cp = A.cp + b * 3;
    bufA[160 + t] = A.cp_w[t * 3] * cp[0] + A.cp_w[t * 3 + 1] * cp[1]
                  + A.cp_w[t * 3 + 2] * cp[2] + A.cp_b[t];
  }
  __syncthreads();
  if (t < 128) {
    const float* w = A.w1 + (size_t)t * 192;
    float acc = A.b1[t];
    for (int c = 0; c < 192; ++c) acc += w[c] * bufA[c];
    bufB[t] = acc > 0.f ? acc : 0.01f * acc;       // leaky_relu
  }
  __syncthreads();
  if (t < 128) {
    const float* w = A.w2 + (size_t)t * 128;
    float acc = A.b2[t];
    for (int c = 0; c < 128; ++c) acc += w[c] * bufB[c];
    bufA[t] = acc > 0.f ? acc : 0.f;
  }
  __syncthreads();
  if (t == 0) {
    float acc = A.b3[0];
    for (int c = 0; c < 128; ++c) acc += A.w3[c] * bufA[c];
    A.out[b] = 1.f / (1.f + expf(-acc));
  }
}

// ---------------------------------------------------------------------------
// Host orchestration
// ---------------------------------------------------------------------------
static inline int pad32(int c) { return (c + 31) & ~31; }

extern "C" void kernel_launch(void* const* d_in, const int* in_sizes, int n_in,
                              void* d_out, int out_size, void* d_ws, size_t ws_size,
                              hipStream_t stream) {
  (void)in_sizes; (void)n_in; (void)out_size; (void)ws_size;
  char* wsb = (char*)d_ws;
  size_t cur = 0;
  auto alloc = [&](size_t bytes) -> size_t {
    size_t p = cur;
    cur += (bytes + 255) & ~(size_t)255;
    return p;
  };
  auto P = [&](size_t off) -> void* { return wsb + off; };

  const float* pcs  = (const float*)d_in[0];
  const float* task = (const float*)d_in[1];
  const float* cp   = (const float*)d_in[2];

  // ---- network shape tables ----
  const int sa_np[4]   = {1024, 256, 64, 16};
  const float sa_r[4]  = {0.1f, 0.2f, 0.4f, 0.8f};
  const int sa_dims[4][4] = {{6,32,32,64},{67,64,64,128},{131,128,128,256},{259,256,256,512}};
  const int fp_dims[4][3] = {{131,128,128},{320,256,128},{384,256,256},{768,256,256}};
  // d_in bases: sa layer (L,j) -> 3 + (L*3+j)*5 ; fp layer bases:
  const int fp_base[4][2] = {{63,68},{78,83},{88,93},{98,103}}; // fp0 layer2 at 73 (head)

  // ---- weight prep (f16 + folded BN) ----
  size_t saW[4][3], saS[4][3], saH[4][3];
  size_t fpW[4][2], fpS[4][2], fpH[4][2];
  auto prep = [&](int base, int o, int c, size_t& wo, size_t& so, size_t& ho) {
    int kp = pad32(c);
    wo = alloc((size_t)o * kp * 2);
    so = alloc((size_t)o * 4);
    ho = alloc((size_t)o * 4);
    int total = o * kp;
    prep_weights_kernel<<<(total + 255) / 256, 256, 0, stream>>>(
        (const float*)d_in[base], (const float*)d_in[base + 1], (const float*)d_in[base + 2],
        (const float*)d_in[base + 3], (const float*)d_in[base + 4],
        (_Float16*)P(wo), (float*)P(so), (float*)P(ho), o, c, kp);
  };
  for (int L = 0; L < 4; ++L)
    for (int j = 0; j < 3; ++j)
      prep(3 + (L * 3 + j) * 5, sa_dims[L][j + 1], sa_dims[L][j], saW[L][j], saS[L][j], saH[L][j]);
  for (int L = 1; L < 4; ++L) {   // fp levels 1..3 use WMMA; fp level 0 handled in head (f32)
    int d0 = fp_dims[L][0], d1 = fp_dims[L][1], d2 = fp_dims[L][2];
    prep(fp_base[L][0], d1, d0, fpW[L][0], fpS[L][0], fpH[L][0]);
    prep(fp_base[L][1], d2, d1, fpW[L][1], fpS[L][1], fpH[L][1]);
  }

  // ---- activation / index buffers ----
  size_t feat0 = alloc((size_t)NB * NPTS * 3 * 2);
  f32_to_f16_kernel<<<(NB * NPTS * 3 + 255) / 256, 256, 0, stream>>>(
      pcs, (_Float16*)P(feat0), NB * NPTS * 3);

  size_t lx[4], saFeat[4];
  for (int L = 0; L < 4; ++L) {
    lx[L]     = alloc((size_t)NB * sa_np[L] * 3 * 4);
    saFeat[L] = alloc((size_t)NB * sa_np[L] * sa_dims[L][3] * 2);
  }
  size_t fpFeat3 = alloc((size_t)NB * 64 * 256 * 2);
  size_t fpFeat2 = alloc((size_t)NB * 256 * 256 * 2);
  size_t fpFeat1 = alloc((size_t)NB * 1024 * 128 * 2);
  size_t fpsIdx  = alloc((size_t)NB * 1024 * 4);
  size_t ballIdx = alloc((size_t)NB * 1024 * 32 * 4);
  size_t knnIdx  = alloc((size_t)NB * 1024 * 3 * 4);
  size_t knnW    = alloc((size_t)NB * 1024 * 3 * 4);
  size_t slotX   = alloc((size_t)524288 * 32 * 2);   // 33.5 MB
  size_t slotY   = alloc((size_t)524288 * 64 * 2);   // 67 MB

  auto gemm = [&](const void* A, size_t M, int Kpad, int O,
                  size_t wo, size_t so, size_t ho, void* Out) {
    dim3 g((unsigned)(M / 32), (unsigned)((O / 16 + 3) / 4));  // 32-row strips
    gemm_bn_relu_kernel<<<g, 128, 0, stream>>>(
        (const _Float16*)A, Kpad, O, (const _Float16*)P(wo),
        (const float*)P(so), (const float*)P(ho), (_Float16*)Out);
  };

  // ---- SA chain ----
  const float* xyzPrev = pcs;
  const _Float16* featPrev = (const _Float16*)P(feat0);
  int nPrev = NPTS, Cprev = 3;
  for (int L = 0; L < 4; ++L) {
    int m = sa_np[L], ns = 32;
    float r2 = sa_r[L] * sa_r[L];
    fps_kernel<<<NB, 256, 0, stream>>>(xyzPrev, nPrev, m, (int*)P(fpsIdx));
    gather_xyz_kernel<<<(NB * m + 255) / 256, 256, 0, stream>>>(
        xyzPrev, (const int*)P(fpsIdx), nPrev, m, NB * m, (float*)P(lx[L]));
    ball_query_kernel<<<(NB * m + 255) / 256, 256, 0, stream>>>(
        xyzPrev, (const float*)P(lx[L]), nPrev, m, ns, r2, NB * m, (int*)P(ballIdx));
    int rows = NB * m * ns;
    int K0 = pad32(3 + Cprev);
    group_kernel<<<(rows + 255) / 256, 256, 0, stream>>>(
        xyzPrev, (const float*)P(lx[L]), featPrev, Cprev, (const int*)P(ballIdx),
        m, ns, nPrev, K0, rows, (_Float16*)P(slotX));
    gemm(P(slotX), rows, K0,            sa_dims[L][1], saW[L][0], saS[L][0], saH[L][0], P(slotY));
    gemm(P(slotY), rows, sa_dims[L][1], sa_dims[L][2], saW[L][1], saS[L][1], saH[L][1], P(slotX));
    gemm(P(slotX), rows, sa_dims[L][2], sa_dims[L][3], saW[L][2], saS[L][2], saH[L][2], P(slotY));
    int O = sa_dims[L][3];
    maxpool_kernel<<<(NB * m * O + 255) / 256, 256, 0, stream>>>(
        (const _Float16*)P(slotY), ns, O, NB * m * O, (_Float16*)P(saFeat[L]));
    xyzPrev = (const float*)P(lx[L]);
    featPrev = (const _Float16*)P(saFeat[L]);
    nPrev = m; Cprev = O;
  }

  // ---- FP chain (levels 3 -> 1; level 0 handled in head) ----
  auto fpStage = [&](int L, const void* xyz1, const void* xyz2, int n1, int n2,
                     const void* f1, int C1, const void* f2, int C2, void* outFeat) {
    knn3_kernel<<<(NB * n1 + 255) / 256, 256, 0, stream>>>(
        (const float*)xyz1, (const float*)xyz2, n1, n2, NB * n1,
        (int*)P(knnIdx), (float*)P(knnW));
    int Kpad = C1 + C2;   // always multiple of 32 here
    int total = NB * n1 * Kpad;
    fillH_kernel<<<(total + 255) / 256, 256, 0, stream>>>(
        (const int*)P(knnIdx), (const float*)P(knnW),
        (const _Float16*)f2, C2, n2, (const _Float16*)f1, C1, n1, Kpad, total,
        (_Float16*)P(slotX));
    size_t rows = (size_t)NB * n1;
    gemm(P(slotX), rows, Kpad,          fp_dims[L][1], fpW[L][0], fpS[L][0], fpH[L][0], P(slotY));
    gemm(P(slotY), rows, fp_dims[L][1], fp_dims[L][2], fpW[L][1], fpS[L][1], fpH[L][1], outFeat);
  };
  fpStage(3, P(lx[2]), P(lx[3]), 64, 16,   P(saFeat[2]), 256, P(saFeat[3]), 512, P(fpFeat3));
  fpStage(2, P(lx[1]), P(lx[2]), 256, 64,  P(saFeat[1]), 128, P(fpFeat3),   256, P(fpFeat2));
  fpStage(1, P(lx[0]), P(lx[1]), 1024, 256, P(saFeat[0]), 64, P(fpFeat2),   256, P(fpFeat1));

  // ---- head (point 0 only) ----
  HeadArgs HA;
  HA.pcs = pcs; HA.task = task; HA.cp = cp;
  HA.lx1 = (const float*)P(lx[0]);
  HA.f2  = (const _Float16*)P(fpFeat1);
  HA.W0 = (const float*)d_in[63]; HA.ga0 = (const float*)d_in[64]; HA.be0 = (const float*)d_in[65];
  HA.mu0 = (const float*)d_in[66]; HA.va0 = (const float*)d_in[67];
  HA.W1 = (const float*)d_in[68]; HA.ga1 = (const float*)d_in[69]; HA.be1 = (const float*)d_in[70];
  HA.mu1 = (const float*)d_in[71]; HA.va1 = (const float*)d_in[72];
  HA.W2 = (const float*)d_in[73]; HA.ga2 = (const float*)d_in[74]; HA.be2 = (const float*)d_in[75];
  HA.mu2 = (const float*)d_in[76]; HA.va2 = (const float*)d_in[77];
  HA.Wf = (const float*)d_in[108]; HA.gaf = (const float*)d_in[109]; HA.bef = (const float*)d_in[110];
  HA.muf = (const float*)d_in[111]; HA.vaf = (const float*)d_in[112];
  HA.task_w = (const float*)d_in[113]; HA.task_b = (const float*)d_in[114];
  HA.cp_w   = (const float*)d_in[115]; HA.cp_b   = (const float*)d_in[116];
  HA.w1 = (const float*)d_in[117]; HA.b1 = (const float*)d_in[118];
  HA.w2 = (const float*)d_in[119]; HA.b2 = (const float*)d_in[120];
  HA.w3 = (const float*)d_in[121]; HA.b3 = (const float*)d_in[122];
  HA.out = (float*)d_out;
  head_kernel<<<NB, 128, 0, stream>>>(HA);
}
